// AttentionBlock_24618752540866
// MI455X (gfx1250) — matrix operand order repaired and checked by reading, compiled, not run on hardware
//
#include <hip/hip_runtime.h>

// ---------------------------------------------------------------- types ----
typedef __bf16 bf16;
typedef bf16  v16bf __attribute__((ext_vector_type(16)));
typedef bf16  v8bf  __attribute__((ext_vector_type(8)));
typedef float v8f   __attribute__((ext_vector_type(8)));

static __device__ __forceinline__ bf16 tobf(float f) {
  unsigned u = __builtin_bit_cast(unsigned, f);
  unsigned r = u + 0x7FFFu + ((u >> 16) & 1u);   // round-to-nearest-even
  unsigned short h = (unsigned short)(r >> 16);
  return __builtin_bit_cast(bf16, h);
}
static __device__ __forceinline__ float tof(bf16 b) {
  unsigned short h = __builtin_bit_cast(unsigned short, b);
  unsigned u = ((unsigned)h) << 16;
  return __builtin_bit_cast(float, u);
}
static __device__ __forceinline__ v16bf cat16(v8bf lo, v8bf hi) {
  return __builtin_shufflevector(lo, hi, 0,1,2,3,4,5,6,7,8,9,10,11,12,13,14,15);
}
static __device__ __forceinline__ float waveSum(float v) {
#pragma unroll
  for (int m = 16; m > 0; m >>= 1) v += __shfl_xor(v, m, 32);
  return v;
}
// CDNA5 async copy: global -> LDS, 16B per lane, tracked by ASYNCcnt.
static __device__ __forceinline__ void async_g2l_b128(const void* gptr, void* lptr) {
  unsigned lds = (unsigned)(unsigned long long)(uintptr_t)lptr;   // low 32b = LDS offset
  unsigned long long ga = (unsigned long long)(uintptr_t)gptr;
  asm volatile("global_load_async_to_lds_b128 %0, %1, off" :: "v"(lds), "v"(ga) : "memory");
}
static __device__ __forceinline__ void wait_async0() {
  asm volatile("s_wait_asynccnt 0x0" ::: "memory");
}

// ---------------------------------------------------- 1) cast + transpose --
// W [K][Nw] f32 (row-major)  ->  Wt [Nw][K] bf16 (row-major)
__global__ __launch_bounds__(256) void castT_kernel(const float* __restrict__ W,
                                                    bf16* __restrict__ Wt,
                                                    int K, int Nw) {
  __shared__ float tile[32][33];
  int tid = threadIdx.x, tx = tid & 31, ty = tid >> 5;
  int nx = blockIdx.x, ky = blockIdx.y;
#pragma unroll
  for (int i = 0; i < 4; ++i) {
    int kk = ky * 32 + ty + i * 8;
    int nn = nx * 32 + tx;
    tile[ty + i * 8][tx] = W[(size_t)kk * Nw + nn];
  }
  __syncthreads();
#pragma unroll
  for (int i = 0; i < 4; ++i) {
    int nn = nx * 32 + ty + i * 8;
    int kk = ky * 32 + tx;
    Wt[(size_t)nn * K + kk] = tobf(tile[tx][ty + i * 8]);
  }
}

// ----------------------------------------------- 2) scale/shift tiny GEMV --
__global__ __launch_bounds__(256) void scale_shift_kernel(const float* __restrict__ emb,
                                                          const float* __restrict__ W_emb,
                                                          const float* __restrict__ b_emb,
                                                          float* __restrict__ ss) {
  int j = blockIdx.x * 256 + threadIdx.x;   // 0..4095
  int b = blockIdx.y;
  float acc = b_emb[j];
  const float* e = emb + (size_t)b * 2048;
  for (int k = 0; k < 2048; ++k)
    acc = fmaf(e[k], W_emb[(size_t)k * 4096 + j], acc);
  ss[(size_t)b * 4096 + j] = acc;
}

// ----------------------------------------------------------- 3) AdaLN -----
__global__ __launch_bounds__(256) void adaln_kernel(const float* __restrict__ x,
                                                    const float* __restrict__ ss,
                                                    const float* __restrict__ g,
                                                    bf16* __restrict__ xn) {
  __shared__ float red[8];
  int row = blockIdx.x;            // 0..8191
  int b = row >> 11;
  int tid = threadIdx.x;
  const float* xr = x + (size_t)row * 2048;
  float vals[8];
  float s = 0.f;
#pragma unroll
  for (int i = 0; i < 8; ++i) {
    vals[i] = xr[tid + i * 256];
    s = fmaf(vals[i], vals[i], s);
  }
  s = waveSum(s);
  if ((tid & 31) == 0) red[tid >> 5] = s;
  __syncthreads();
  float tot = 0.f;
#pragma unroll
  for (int i = 0; i < 8; ++i) tot += red[i];
  float inv = rsqrtf(tot * (1.0f / 2048.0f) + 1e-6f);
  const float* scp = ss + (size_t)b * 4096;
#pragma unroll
  for (int i = 0; i < 8; ++i) {
    int col = tid + i * 256;
    float y = vals[i] * inv * g[col];
    y = fmaf(y, scp[col], y) + scp[2048 + col];   // y*(1+scale)+shift
    xn[(size_t)row * 2048 + col] = tobf(y);
  }
}

// ------------------------------------------------- 4/7) WMMA bf16 GEMM ----
// C[M][N] = A[M][K](bf16) * Bt[N][K](bf16)^T ; out bf16 (Cb) or f32+res (Cf)
// 128x128 tile/WG, BK=32, double-buffered async-to-LDS staging.
__global__ __launch_bounds__(256) void gemm_bf16_wmma(const bf16* __restrict__ A,
                                                      const bf16* __restrict__ Bt,
                                                      int M, int N, int K,
                                                      bf16* __restrict__ Cb,
                                                      float* __restrict__ Cf,
                                                      const float* __restrict__ res) {
  __shared__ __align__(16) bf16 As[2][128][40];
  __shared__ __align__(16) bf16 Bs[2][128][40];

  int tid = threadIdx.x;
  int wave = tid >> 5, lane = tid & 31, half = lane >> 4, l15 = lane & 15;
  int wM = wave >> 2, wN = wave & 3;          // 2 x 4 wave grid
  int row0 = blockIdx.y * 128, col0 = blockIdx.x * 128;

  v8f zero = {0.f,0.f,0.f,0.f,0.f,0.f,0.f,0.f};
  v8f acc[4][2];
#pragma unroll
  for (int i = 0; i < 4; ++i)
#pragma unroll
    for (int j = 0; j < 2; ++j) acc[i][j] = zero;

  // async DMA of one 128x32 A tile + B tile into buffer bb
  auto issueTile = [&](int bb, int kb) {
#pragma unroll
    for (int c = 0; c < 2; ++c) {
      int idx = c * 256 + tid;                // 512 x 16B per matrix
      int row = idx >> 2, seg = idx & 3;
      async_g2l_b128(A + (size_t)(row0 + row) * K + kb + seg * 8, &As[bb][row][seg * 8]);
      async_g2l_b128(Bt + (size_t)(col0 + row) * K + kb + seg * 8, &Bs[bb][row][seg * 8]);
    }
  };

  issueTile(0, 0);
  int buf = 0;
  for (int kb = 0; kb < K; kb += 32) {
    wait_async0();            // my async loads for `buf` are done
    __syncthreads();          // everyone's are done; prev reads of buf^1 done
    if (kb + 32 < K) issueTile(buf ^ 1, kb + 32);   // overlap DMA with WMMAs
    if (kb + 64 < K) {
      __builtin_prefetch(A + (size_t)(row0 + (tid >> 1)) * K + kb + 64 + (tid & 1) * 16, 0, 1);
      __builtin_prefetch(Bt + (size_t)(col0 + (tid >> 1)) * K + kb + 64 + (tid & 1) * 16, 0, 1);
    }

    v16bf bfrag[2];
#pragma unroll
    for (int j = 0; j < 2; ++j) {
      int nn = wN * 32 + j * 16 + l15;
      int k0 = half * 16;                      // B: 16 contiguous K per lane
      v8bf lo = *(const v8bf*)&Bs[buf][nn][k0];
      v8bf hi = *(const v8bf*)&Bs[buf][nn][k0 + 8];
      bfrag[j] = cat16(lo, hi);
    }
#pragma unroll
    for (int i = 0; i < 4; ++i) {
      int mm = wM * 64 + i * 16 + l15;
      int k0 = half * 16;                      // A: 16 contiguous K per lane
      v8bf lo = *(const v8bf*)&As[buf][mm][k0];
      v8bf hi = *(const v8bf*)&As[buf][mm][k0 + 8];
      v16bf af = cat16(lo, hi);
#pragma unroll
      for (int j = 0; j < 2; ++j)
        acc[i][j] = __builtin_amdgcn_wmma_f32_16x16x32_bf16(
            false, af, false, bfrag[j], (short)0, acc[i][j], false, false);
    }
    buf ^= 1;
  }

#pragma unroll
  for (int i = 0; i < 4; ++i)
#pragma unroll
    for (int j = 0; j < 2; ++j)
#pragma unroll
      for (int r = 0; r < 8; ++r) {
        int gm = row0 + wM * 64 + i * 16 + r + 8 * half;
        int gn = col0 + wN * 32 + j * 16 + l15;
        size_t idx = (size_t)gm * N + gn;
        float vv = acc[i][j][r];
        if (Cf) Cf[idx] = vv + (res ? res[idx] : 0.f);
        else    Cb[idx] = tobf(vv);
      }
}

// ------------------------------------- 5) per-head q/k RMSNorm + split ----
__global__ __launch_bounds__(128) void qk_norm_split(const bf16* __restrict__ qkv,
                                                     const float* __restrict__ gq,
                                                     const float* __restrict__ gk,
                                                     bf16* __restrict__ qa,
                                                     bf16* __restrict__ ka,
                                                     bf16* __restrict__ va) {
  __shared__ float red[4];
  int row = blockIdx.x;               // 0..8191  (b*2048+n)
  int b = row >> 11, n = row & 2047;
  int t = threadIdx.x;                // 0..127
  const bf16* rp = qkv + (size_t)row * 6144;
  for (int h = 0; h < 16; ++h) {
    size_t dst = ((size_t)(b * 16 + h) * 2048 + n) * 128 + t;
    // q
    float qv = tof(rp[h * 128 + t]);
    float sq = waveSum(qv * qv);
    if ((t & 31) == 0) red[t >> 5] = sq;
    __syncthreads();
    float tq = red[0] + red[1] + red[2] + red[3];
    __syncthreads();
    qa[dst] = tobf(qv * rsqrtf(tq * (1.0f / 128.0f) + 1e-6f) * gq[t]);
    // k
    float kv = tof(rp[2048 + h * 128 + t]);
    float sk = waveSum(kv * kv);
    if ((t & 31) == 0) red[t >> 5] = sk;
    __syncthreads();
    float tk = red[0] + red[1] + red[2] + red[3];
    __syncthreads();
    ka[dst] = tobf(kv * rsqrtf(tk * (1.0f / 128.0f) + 1e-6f) * gk[t]);
    // v passthrough
    va[dst] = rp[4096 + h * 128 + t];
  }
}

// ------------------------------------------- 6) flash attention (WMMA) ----
__global__ __launch_bounds__(256) void attn_kernel(const bf16* __restrict__ q,
                                                   const bf16* __restrict__ k,
                                                   const bf16* __restrict__ v,
                                                   bf16* __restrict__ ob,
                                                   float* __restrict__ of) {
  const int NSEQ = 2048, D = 128;
  __shared__ __align__(16) bf16 Ks[32][136];     // [key][d]
  __shared__ __align__(16) bf16 Vt[128][40];     // [d][key]   (transposed)
  __shared__ __align__(16) bf16 Ps[8][16][40];   // per-wave P transpose

  int tid = threadIdx.x;
  int wave = tid >> 5, lane = tid & 31, half = lane >> 4, l15 = lane & 15;
  int bh = blockIdx.y;                           // b*16+h
  int b = bh >> 4, h = bh & 15;
  int qBase = blockIdx.x * 128 + wave * 16;

  const bf16* qp = q + (size_t)bh * NSEQ * D;
  const bf16* kp = k + (size_t)bh * NSEQ * D;
  const bf16* vp = v + (size_t)bh * NSEQ * D;

  // Q fragments: A-layout, 4 chunks of K(=d)=32
  v16bf Qf[4];
#pragma unroll
  for (int c = 0; c < 4; ++c) {
    const bf16* base = qp + (size_t)(qBase + l15) * D + c * 32 + half * 16;
    Qf[c] = cat16(*(const v8bf*)base, *(const v8bf*)(base + 8));
  }

  v8f zero = {0.f,0.f,0.f,0.f,0.f,0.f,0.f,0.f};
  float mrow[8], lrow[8];
  v8f Oacc[8];
#pragma unroll
  for (int r = 0; r < 8; ++r) { mrow[r] = -1e30f; lrow[r] = 0.f; }
#pragma unroll
  for (int dc = 0; dc < 8; ++dc) Oacc[dc] = zero;

  const float SCALE = 0.08838834764831845f;      // 1/sqrt(128)

  for (int kb = 0; kb < NSEQ; kb += 32) {
    // stage K block [32][128] via async DMA to LDS
#pragma unroll
    for (int c = 0; c < 2; ++c) {
      int idx = c * 256 + tid;                   // 512 x 16B = 8KB
      int row = idx >> 4, seg = idx & 15;
      async_g2l_b128(kp + (size_t)(kb + row) * D + seg * 8, &Ks[row][seg * 8]);
    }
    // stage V block transposed -> Vt[d][key]  (needs VGPR pass-through)
    {
      int key = tid >> 3, dc = tid & 7;
      const bf16* src = vp + (size_t)(kb + key) * D + dc * 16;
      v8bf a = *(const v8bf*)src;
      v8bf bb = *(const v8bf*)(src + 8);
#pragma unroll
      for (int i = 0; i < 8; ++i) {
        Vt[dc * 16 + i][key] = a[i];
        Vt[dc * 16 + 8 + i][key] = bb[i];
      }
    }
    wait_async0();
    __syncthreads();

    // S = Q * K^T  (16 x 32), two 16-key subtiles
    v8f S0 = zero, S1 = zero;
#pragma unroll
    for (int c = 0; c < 4; ++c) {
      int k0 = c * 32 + half * 16;
      v8bf lo0 = *(const v8bf*)&Ks[l15][k0];
      v8bf hi0 = *(const v8bf*)&Ks[l15][k0 + 8];
      S0 = __builtin_amdgcn_wmma_f32_16x16x32_bf16(false, Qf[c], false, cat16(lo0, hi0),
                                                   (short)0, S0, false, false);
      v8bf lo1 = *(const v8bf*)&Ks[16 + l15][k0];
      v8bf hi1 = *(const v8bf*)&Ks[16 + l15][k0 + 8];
      S1 = __builtin_amdgcn_wmma_f32_16x16x32_bf16(false, Qf[c], false, cat16(lo1, hi1),
                                                   (short)0, S1, false, false);
    }

    // online softmax (row reductions across 16-lane halves)
    float p0[8], p1[8], alpha[8];
#pragma unroll
    for (int r = 0; r < 8; ++r) {
      float s0 = S0[r] * SCALE, s1 = S1[r] * SCALE;
      float mx = fmaxf(s0, s1);
#pragma unroll
      for (int off = 8; off > 0; off >>= 1) mx = fmaxf(mx, __shfl_xor(mx, off, 32));
      float mn = fmaxf(mrow[r], mx);
      alpha[r] = __expf(mrow[r] - mn);
      p0[r] = __expf(s0 - mn);
      p1[r] = __expf(s1 - mn);
      float sum = p0[r] + p1[r];
#pragma unroll
      for (int off = 8; off > 0; off >>= 1) sum += __shfl_xor(sum, off, 32);
      lrow[r] = lrow[r] * alpha[r] + sum;
      mrow[r] = mn;
    }
#pragma unroll
    for (int dc = 0; dc < 8; ++dc)
#pragma unroll
      for (int r = 0; r < 8; ++r) Oacc[dc][r] *= alpha[r];

    // transpose P through per-wave LDS: C-layout -> A-layout
#pragma unroll
    for (int r = 0; r < 8; ++r) {
      int prow = r + 8 * half;
      Ps[wave][prow][l15]      = tobf(p0[r]);
      Ps[wave][prow][16 + l15] = tobf(p1[r]);
    }
    asm volatile("s_wait_dscnt 0x0" ::: "memory");  // LDS store->load hazard
    v16bf Pf;
    {
      int k0 = half * 16;
      v8bf lo = *(const v8bf*)&Ps[wave][l15][k0];
      v8bf hi = *(const v8bf*)&Ps[wave][l15][k0 + 8];
      Pf = cat16(lo, hi);
    }
    // O += P * V
#pragma unroll
    for (int dc = 0; dc < 8; ++dc) {
      int dd = dc * 16 + l15, kk = half * 16;
      v8bf lo = *(const v8bf*)&Vt[dd][kk];
      v8bf hi = *(const v8bf*)&Vt[dd][kk + 8];
      Oacc[dc] = __builtin_amdgcn_wmma_f32_16x16x32_bf16(false, Pf, false, cat16(lo, hi),
                                                         (short)0, Oacc[dc], false, false);
    }
    __syncthreads();
  }

  // epilogue: normalize, write o to [b][n][h*128+d] in f32 + bf16
#pragma unroll
  for (int dc = 0; dc < 8; ++dc)
#pragma unroll
    for (int r = 0; r < 8; ++r) {
      float val = Oacc[dc][r] / lrow[r];
      size_t orow = (size_t)b * NSEQ + qBase + r + 8 * half;
      size_t oidx = orow * 2048 + h * 128 + dc * 16 + l15;
      of[oidx] = val;
      ob[oidx] = tobf(val);
    }
}

// ------------------------------------------------------------- launcher ---
extern "C" void kernel_launch(void* const* d_in, const int* in_sizes, int n_in,
                              void* d_out, int out_size, void* d_ws, size_t ws_size,
                              hipStream_t stream) {
  (void)in_sizes; (void)n_in; (void)out_size; (void)ws_size;
  const float* x      = (const float*)d_in[0];
  const float* emb    = (const float*)d_in[1];
  const float* W_emb  = (const float*)d_in[2];
  const float* b_emb  = (const float*)d_in[3];
  const float* g_norm = (const float*)d_in[4];
  const float* W_qkv  = (const float*)d_in[5];
  const float* g_q    = (const float*)d_in[6];
  const float* g_k    = (const float*)d_in[7];
  const float* W_out  = (const float*)d_in[8];
  float* out = (float*)d_out;

  const int Bb = 4, Nseq = 2048, DIM = 2048, H = 16;
  const int BN = Bb * Nseq;                         // 8192

  char* w = (char*)d_ws;
  auto alloc = [&](size_t bytes) -> char* {
    char* p = w; w += (bytes + 255) & ~(size_t)255; return p;
  };
  bf16*  xn     = (bf16*)alloc((size_t)BN * DIM * 2);
  bf16*  Wqkv_t = (bf16*)alloc((size_t)3 * DIM * DIM * 2);
  bf16*  Wout_t = (bf16*)alloc((size_t)DIM * DIM * 2);
  float* ss     = (float*)alloc((size_t)Bb * 2 * DIM * 4);
  bf16*  qkvb   = (bf16*)alloc((size_t)BN * 3 * DIM * 2);
  bf16*  qa     = (bf16*)alloc((size_t)BN * DIM * 2);
  bf16*  ka     = (bf16*)alloc((size_t)BN * DIM * 2);
  bf16*  va     = (bf16*)alloc((size_t)BN * DIM * 2);
  bf16*  obuf   = (bf16*)alloc((size_t)BN * DIM * 2);
  float* ofl    = (float*)alloc((size_t)BN * DIM * 4);

  // 1) weight casts (fp32 [K][N] -> bf16 [N][K])
  castT_kernel<<<dim3(3 * DIM / 32, DIM / 32), 256, 0, stream>>>(W_qkv, Wqkv_t, DIM, 3 * DIM);
  castT_kernel<<<dim3(DIM / 32, DIM / 32), 256, 0, stream>>>(W_out, Wout_t, DIM, DIM);
  // 2) scale/shift
  scale_shift_kernel<<<dim3(2 * DIM / 256, Bb), 256, 0, stream>>>(emb, W_emb, b_emb, ss);
  // 3) AdaLN
  adaln_kernel<<<dim3(BN), 256, 0, stream>>>(x, ss, g_norm, xn);
  // 4) QKV GEMM (8192 x 6144 x 2048), bf16 out
  gemm_bf16_wmma<<<dim3(3 * DIM / 128, BN / 128), 256, 0, stream>>>(
      xn, Wqkv_t, BN, 3 * DIM, DIM, qkvb, nullptr, nullptr);
  // 5) per-head q/k RMSNorm + head split
  qk_norm_split<<<dim3(BN), 128, 0, stream>>>(qkvb, g_q, g_k, qa, ka, va);
  // 6) attention
  attn_kernel<<<dim3(Nseq / 128, Bb * H), 256, 0, stream>>>(qa, ka, va, obuf, ofl);
  // 7) out projection + residual (f32 out)
  gemm_bf16_wmma<<<dim3(DIM / 128, BN / 128), 256, 0, stream>>>(
      obuf, Wout_t, BN, DIM, DIM, nullptr, out, ofl);
}
